// AnchorBasedTrajectoryDecoder_45518063403476
// MI455X (gfx1250) — matrix operand order repaired and checked by reading, not compiled, not run on hardware
//
#include <hip/hip_runtime.h>
#include <hip/hip_bf16.h>

// ---------------------------------------------------------------------------
// AnchorBasedTrajectoryDecoder for MI455X (gfx1250, wave32, WMMA bf16)
// ---------------------------------------------------------------------------
// N = 6144 rows (BS*K), H = 256, 4H = 1024, T = 30.
// All GEMMs use v_wmma_f32_16x16x32_bf16. Weights pre-swizzled to B-fragment
// layout; recurrent weights gate-interleaved (n' = j*4 + gate) so LSTM
// activations need only intra-wave shuffles. b_hh0 folded into gx0; b_ih1+b_hh1
// pre-summed. Weight prefetch at WGP scope (populates all cache levels).
// ---------------------------------------------------------------------------

typedef __attribute__((ext_vector_type(16))) __bf16 v16bf;
typedef __attribute__((ext_vector_type(8)))  float  v8f;
typedef __attribute__((ext_vector_type(4)))  unsigned int uint4v;

#define WMMA_BF16(A, B, C) \
  __builtin_amdgcn_wmma_f32_16x16x32_bf16(false, (A), false, (B), (short)0, (C), false, false)

union FragU { v16bf v; uint4v u[2]; };

__device__ __forceinline__ float sigmoidf_(float x) {
  return 1.0f / (1.0f + __expf(-x));
}

// B-fragment: frag index (nt*8+kc), 512 bf16 per frag, lane slot = 16 bf16.
__device__ __forceinline__ v16bf load_bfrag(const __bf16* base, int nt, int kc, int lane) {
  const __bf16* p = base + (((size_t)(nt * 8 + kc)) << 9) + (lane << 4);
  FragU f;
  f.u[0] = *(const uint4v*)(p);
  f.u[1] = *(const uint4v*)(p + 8);
  // Next K-chunk; locality 3 -> WGP scope: pulls into all cache levels.
  __builtin_prefetch(p + 512, 0, 3);
  return f.v;
}

// A-fragment from row-major bf16 LDS row: K in {c0..c0+7} U {c0+16..c0+23}.
__device__ __forceinline__ v16bf load_afrag(const __bf16* hrow, int col0) {
  FragU f;
  f.u[0] = *(const uint4v*)(hrow + col0);
  f.u[1] = *(const uint4v*)(hrow + col0 + 16);
  return f.v;
}

// ---------------------------------------------------------------------------
// Workspace layout
// ---------------------------------------------------------------------------
static const size_t OFF_WHID = 0;                       // bf16 elems: 16*8*512
static const size_t OFF_WIH0 = 65536;                   // 64*8*512
static const size_t OFF_WHH0 = 65536 + 262144;          // 327680
static const size_t OFF_WIH1 = 327680 + 262144;         // 589824
static const size_t OFF_WHH1 = 589824 + 262144;         // 851968
static const size_t OFF_BSUM1_BYTES = (851968 + 262144) * 2;      // 2228224: f32[1024]
static const size_t OFF_GX0_BYTES   = OFF_BSUM1_BYTES + 4096;     // 2232320: f32[6144*1024]

// ---------------------------------------------------------------------------
// Kernel A: weight -> bf16 WMMA-B-fragment swizzle (+ bias pre-sum, m==5)
// ---------------------------------------------------------------------------
__global__ __launch_bounds__(256) void prep_frags(
    const float* __restrict__ Whid, const float* __restrict__ Wih0,
    const float* __restrict__ Whh0, const float* __restrict__ Wih1,
    const float* __restrict__ Whh1,
    const float* __restrict__ b_ih1, const float* __restrict__ b_hh1,
    __bf16* __restrict__ ws) {
  int m = blockIdx.y;
  if (m == 5) {   // bsum1 = b_ih1 + b_hh1 (f32)
    int e = blockIdx.x * 256 + threadIdx.x;
    if (e < 1024) {
      float* bsum1 = (float*)((char*)ws + OFF_BSUM1_BYTES);
      bsum1[e] = b_ih1[e] + b_hh1[e];
    }
    return;
  }
  const float* src;
  __bf16* dst;
  int stride, permute, elems;
  switch (m) {
    case 0:  src = Whid; dst = ws + OFF_WHID; stride = 256; permute = 0; elems = 65536;  break;
    case 1:  src = Wih0; dst = ws + OFF_WIH0; stride = 258; permute = 0; elems = 262144; break;
    case 2:  src = Whh0; dst = ws + OFF_WHH0; stride = 256; permute = 1; elems = 262144; break;
    case 3:  src = Wih1; dst = ws + OFF_WIH1; stride = 256; permute = 1; elems = 262144; break;
    default: src = Whh1; dst = ws + OFF_WHH1; stride = 256; permute = 1; elems = 262144; break;
  }
  int e = blockIdx.x * 256 + threadIdx.x;
  if (e >= elems) return;
  int f    = e >> 9;          // fragment
  int lane = (e >> 4) & 31;
  int s    = e & 15;
  int nt = f >> 3, kc = f & 7;
  int N2 = nt * 16 + (lane & 15);
  int K  = kc * 32 + ((s & 8) << 1) + ((lane >> 4) << 3) + (s & 7);
  int N  = permute ? ((N2 & 3) * 256 + (N2 >> 2)) : N2;  // gate-interleave
  dst[e] = (__bf16)src[(size_t)N * stride + K];
}

// ---------------------------------------------------------------------------
// Kernel B: front-end. 192 WGs x 32 rows, 256 threads (8 waves).
// ---------------------------------------------------------------------------
__global__ __launch_bounds__(256) void frontend(
    const float* __restrict__ obs, const float* __restrict__ lanef,
    const float* __restrict__ dstyle,
    const float* __restrict__ b_hid,
    const float* __restrict__ W_ep, const float* __restrict__ b_ep,
    const float* __restrict__ W_conf, const float* __restrict__ b_conf,
    const float* __restrict__ W_ih0_raw, const float* __restrict__ b_ih0,
    const float* __restrict__ b_hh0,
    const __bf16* __restrict__ WhidF, const __bf16* __restrict__ Wih0F,
    float* __restrict__ gx0, float* __restrict__ outConf, float* __restrict__ outEp) {
  __shared__ __bf16 combA[32][256];
  __shared__ __bf16 hidA[32][256];
  __shared__ float epl[32][2];

  const int tid = threadIdx.x, lane = tid & 31, w = tid >> 5;
  const int grow0 = blockIdx.x * 32;
  const int hi8 = (lane & 16) >> 1;

  // Phase 1: build combined = [obs | lane | driving_style] as bf16
  for (int idx = tid; idx < 32 * 256; idx += 256) {
    int row = idx >> 8, col = idx & 255;
    int g = grow0 + row, b = g / 6, kk = g % 6;
    float v = (col < 128) ? obs[b * 128 + col]
            : (col < 192) ? lanef[b * 64 + (col - 128)]
                          : dstyle[kk * 64 + (col - 192)];
    combA[row][col] = (__bf16)v;
  }
  __syncthreads();

  // Phase 2: hidden = leaky_relu(combined @ W_hid^T + b_hid)
  {
    v8f z = {};
    v8f acc[2][2];
    acc[0][0] = z; acc[0][1] = z; acc[1][0] = z; acc[1][1] = z;
    for (int kc = 0; kc < 8; ++kc) {
      int col0 = kc * 32 + ((lane >> 4) << 3);
      v16bf a0 = load_afrag(&combA[lane & 15][0], col0);
      v16bf a1 = load_afrag(&combA[16 + (lane & 15)][0], col0);
#pragma unroll
      for (int tt = 0; tt < 2; ++tt) {
        v16bf bf = load_bfrag(WhidF, w * 2 + tt, kc, lane);
        acc[0][tt] = WMMA_BF16(a0, bf, acc[0][tt]);
        acc[1][tt] = WMMA_BF16(a1, bf, acc[1][tt]);
      }
    }
#pragma unroll
    for (int mt = 0; mt < 2; ++mt)
#pragma unroll
      for (int tt = 0; tt < 2; ++tt)
#pragma unroll
        for (int r = 0; r < 8; ++r) {
          int row = mt * 16 + r + hi8;
          int col = (w * 2 + tt) * 16 + (lane & 15);
          float v = acc[mt][tt][r] + b_hid[col];
          v = v > 0.f ? v : 0.1f * v;
          hidA[row][col] = (__bf16)v;
        }
  }
  __syncthreads();

  // Phase 3: endpoint (2) + conf (1) heads
  if (tid < 96) {
    int row = tid / 3, o = tid % 3;
    const float* wv = (o == 0) ? W_ep : (o == 1) ? (W_ep + 256) : W_conf;
    float s = (o == 0) ? b_ep[0] : (o == 1) ? b_ep[1] : b_conf[0];
    for (int col = 0; col < 256; ++col) s += (float)hidA[row][col] * wv[col];
    int g = grow0 + row;
    if (o < 2) { epl[row][o] = s; outEp[g * 2 + o] = s; }
    else       { outConf[g] = s; }
  }
  __syncthreads();

  // Phase 4: gx0 = [hidden, endpoint] @ W_ih0^T + b_ih0 + b_hh0
  // (b_hh0 folded in here: gx0 is added to every step's layer-0 gates anyway)
  {
    v8f z = {};
    v8f acc[2][8];
#pragma unroll
    for (int tt = 0; tt < 8; ++tt) { acc[0][tt] = z; acc[1][tt] = z; }
    for (int kc = 0; kc < 8; ++kc) {
      int col0 = kc * 32 + ((lane >> 4) << 3);
      v16bf a0 = load_afrag(&hidA[lane & 15][0], col0);
      v16bf a1 = load_afrag(&hidA[16 + (lane & 15)][0], col0);
#pragma unroll
      for (int tt = 0; tt < 8; ++tt) {
        v16bf bf = load_bfrag(Wih0F, w * 8 + tt, kc, lane);
        acc[0][tt] = WMMA_BF16(a0, bf, acc[0][tt]);
        acc[1][tt] = WMMA_BF16(a1, bf, acc[1][tt]);
      }
    }
#pragma unroll
    for (int mt = 0; mt < 2; ++mt)
#pragma unroll
      for (int tt = 0; tt < 8; ++tt)
#pragma unroll
        for (int r = 0; r < 8; ++r) {
          int row = mt * 16 + r + hi8;
          int n = (w * 8 + tt) * 16 + (lane & 15);
          float v = acc[mt][tt][r] + b_ih0[n] + b_hh0[n]
                  + epl[row][0] * W_ih0_raw[n * 258 + 256]
                  + epl[row][1] * W_ih0_raw[n * 258 + 257];
          gx0[(size_t)(grow0 + row) * 1024 + n] = v;
        }
  }
}

// ---------------------------------------------------------------------------
// Kernel C: persistent fused 2-layer LSTM decoder, 30 steps.
// 192 WGs x 32 rows; h-states in LDS (bf16 = WMMA A source), c-states in regs.
// Gate-interleaved weight columns => activations via intra-wave __shfl only.
// ---------------------------------------------------------------------------
__global__ __launch_bounds__(256) void decoder(
    const float* __restrict__ gx0,
    const float* __restrict__ bsum1,
    const float* __restrict__ W_op, const float* __restrict__ b_op,
    const __bf16* __restrict__ Whh0F, const __bf16* __restrict__ Wih1F,
    const __bf16* __restrict__ Whh1F,
    float* __restrict__ outFinal) {
  __shared__ __bf16 hA0[32][256];   // LSTM0 hidden state (bf16, A-matrix)
  __shared__ __bf16 hA1[32][256];   // LSTM1 hidden state

  const int tid = threadIdx.x, lane = tid & 31, w = tid >> 5;
  const int grow0 = blockIdx.x * 32;
  const int hi8 = (lane & 16) >> 1;
  const int jsel = lane & 3;
  const int jbase = w * 32;
  const int sbase = (lane & 16) | (jsel << 2);
  const bool writer = (lane & 15) < 4;

  for (int idx = tid; idx < 8192; idx += 256) {
    (&hA0[0][0])[idx] = (__bf16)0.0f;
    (&hA1[0][0])[idx] = (__bf16)0.0f;
  }
  __syncthreads();

  float c0[2][8][8], c1[2][8][8];
#pragma unroll
  for (int mt = 0; mt < 2; ++mt)
#pragma unroll
    for (int tt = 0; tt < 8; ++tt)
#pragma unroll
      for (int r = 0; r < 8; ++r) { c0[mt][tt][r] = 0.f; c1[mt][tt][r] = 0.f; }

  v8f z = {};

  for (int t = 0; t < 30; ++t) {
    // ---- MM0: h0 @ W_hh0^T (gate-interleaved columns) ----
    v8f acc[2][8];
#pragma unroll
    for (int tt = 0; tt < 8; ++tt) { acc[0][tt] = z; acc[1][tt] = z; }
    for (int kc = 0; kc < 8; ++kc) {
      int col0 = kc * 32 + ((lane >> 4) << 3);
      v16bf a0 = load_afrag(&hA0[lane & 15][0], col0);
      v16bf a1 = load_afrag(&hA0[16 + (lane & 15)][0], col0);
#pragma unroll
      for (int tt = 0; tt < 8; ++tt) {
        v16bf bf = load_bfrag(Whh0F, w * 8 + tt, kc, lane);
        acc[0][tt] = WMMA_BF16(a0, bf, acc[0][tt]);
        acc[1][tt] = WMMA_BF16(a1, bf, acc[1][tt]);
      }
    }
    __syncthreads();

    // ---- ACT0: gates = gx0(+biases) + D; update c0, write h0 ----
#pragma unroll
    for (int mt = 0; mt < 2; ++mt)
#pragma unroll
      for (int tt = 0; tt < 8; ++tt) {
        int jg = jbase + tt * 4 + jsel;
#pragma unroll
        for (int r = 0; r < 8; ++r) {
          float dv = acc[mt][tt][r];
          float pi = __shfl(dv, sbase + 0, 32);
          float pf = __shfl(dv, sbase + 1, 32);
          float pg = __shfl(dv, sbase + 2, 32);
          float po = __shfl(dv, sbase + 3, 32);
          int row = mt * 16 + r + hi8;
          const float* gx = gx0 + (size_t)(grow0 + row) * 1024;
          pi += gx[jg];
          pf += gx[256 + jg];
          pg += gx[512 + jg];
          po += gx[768 + jg];
          float cn = sigmoidf_(pf) * c0[mt][tt][r] + sigmoidf_(pi) * tanhf(pg);
          float hn = sigmoidf_(po) * tanhf(cn);
          c0[mt][tt][r] = cn;
          if (writer) hA0[row][jg] = (__bf16)hn;
        }
      }
    __syncthreads();

    // ---- MM1: h0_new @ W_ih1^T + h1_prev @ W_hh1^T ----
#pragma unroll
    for (int tt = 0; tt < 8; ++tt) { acc[0][tt] = z; acc[1][tt] = z; }
    for (int kc = 0; kc < 8; ++kc) {
      int col0 = kc * 32 + ((lane >> 4) << 3);
      v16bf a0 = load_afrag(&hA0[lane & 15][0], col0);
      v16bf a1 = load_afrag(&hA0[16 + (lane & 15)][0], col0);
#pragma unroll
      for (int tt = 0; tt < 8; ++tt) {
        v16bf bf = load_bfrag(Wih1F, w * 8 + tt, kc, lane);
        acc[0][tt] = WMMA_BF16(a0, bf, acc[0][tt]);
        acc[1][tt] = WMMA_BF16(a1, bf, acc[1][tt]);
      }
    }
    for (int kc = 0; kc < 8; ++kc) {
      int col0 = kc * 32 + ((lane >> 4) << 3);
      v16bf a0 = load_afrag(&hA1[lane & 15][0], col0);
      v16bf a1 = load_afrag(&hA1[16 + (lane & 15)][0], col0);
#pragma unroll
      for (int tt = 0; tt < 8; ++tt) {
        v16bf bf = load_bfrag(Whh1F, w * 8 + tt, kc, lane);
        acc[0][tt] = WMMA_BF16(a0, bf, acc[0][tt]);
        acc[1][tt] = WMMA_BF16(a1, bf, acc[1][tt]);
      }
    }
    __syncthreads();

    // ---- ACT1: gates = D + bsum1; update c1, write h1 ----
#pragma unroll
    for (int mt = 0; mt < 2; ++mt)
#pragma unroll
      for (int tt = 0; tt < 8; ++tt) {
        int jg = jbase + tt * 4 + jsel;
        float bi = bsum1[jg];
        float bff = bsum1[256 + jg];
        float bg = bsum1[512 + jg];
        float bo = bsum1[768 + jg];
#pragma unroll
        for (int r = 0; r < 8; ++r) {
          float dv = acc[mt][tt][r];
          float pi = __shfl(dv, sbase + 0, 32) + bi;
          float pf = __shfl(dv, sbase + 1, 32) + bff;
          float pg = __shfl(dv, sbase + 2, 32) + bg;
          float po = __shfl(dv, sbase + 3, 32) + bo;
          int row = mt * 16 + r + hi8;
          float cn = sigmoidf_(pf) * c1[mt][tt][r] + sigmoidf_(pi) * tanhf(pg);
          float hn = sigmoidf_(po) * tanhf(cn);
          c1[mt][tt][r] = cn;
          if (writer) hA1[row][jg] = (__bf16)hn;
        }
      }
    __syncthreads();

    // ---- OUT: final[:, t, :] = h1 @ W_op^T + b_op ----
    if (tid < 64) {
      int row = tid >> 1, jo = tid & 1;
      const float* wrow = W_op + jo * 256;
      float s = b_op[jo];
      for (int col = 0; col < 256; ++col) s += (float)hA1[row][col] * wrow[col];
      outFinal[(size_t)(grow0 + row) * 60 + t * 2 + jo] = s;
    }
    __syncthreads();
  }
}

// ---------------------------------------------------------------------------
// Launch
// ---------------------------------------------------------------------------
extern "C" void kernel_launch(void* const* d_in, const int* in_sizes, int n_in,
                              void* d_out, int out_size, void* d_ws, size_t ws_size,
                              hipStream_t stream) {
  const float* obs     = (const float*)d_in[0];
  const float* lanef   = (const float*)d_in[1];
  const float* dstyle  = (const float*)d_in[3];
  const float* W_hid   = (const float*)d_in[4];
  const float* b_hid   = (const float*)d_in[5];
  const float* W_ep    = (const float*)d_in[6];
  const float* b_ep    = (const float*)d_in[7];
  const float* W_conf  = (const float*)d_in[8];
  const float* b_conf  = (const float*)d_in[9];
  const float* W_ih0   = (const float*)d_in[10];
  const float* W_hh0   = (const float*)d_in[11];
  const float* b_ih0   = (const float*)d_in[12];
  const float* b_hh0   = (const float*)d_in[13];
  const float* W_ih1   = (const float*)d_in[14];
  const float* W_hh1   = (const float*)d_in[15];
  const float* b_ih1   = (const float*)d_in[16];
  const float* b_hh1   = (const float*)d_in[17];
  const float* W_op    = (const float*)d_in[18];
  const float* b_op    = (const float*)d_in[19];

  __bf16* wsb  = (__bf16*)d_ws;
  const float* bsum1 = (const float*)((char*)d_ws + OFF_BSUM1_BYTES);
  float* gx0 = (float*)((char*)d_ws + OFF_GX0_BYTES);

  float* outFinal = (float*)d_out;             // [1024,6,30,2] = 368640
  float* outConf  = outFinal + 368640;         // [1024,6]      =   6144
  float* outEp    = outFinal + 374784;         // [1024,6,2]    =  12288

  prep_frags<<<dim3(1024, 6), 256, 0, stream>>>(W_hid, W_ih0, W_hh0, W_ih1, W_hh1,
                                                b_ih1, b_hh1, wsb);

  frontend<<<192, 256, 0, stream>>>(obs, lanef, dstyle, b_hid, W_ep, b_ep,
                                    W_conf, b_conf, W_ih0, b_ih0, b_hh0,
                                    wsb + OFF_WHID, wsb + OFF_WIH0,
                                    gx0, outConf, outEp);

  decoder<<<192, 256, 0, stream>>>(gx0, bsum1, W_op, b_op,
                                   wsb + OFF_WHH0, wsb + OFF_WIH1, wsb + OFF_WHH1,
                                   outFinal);
}
